// MLPConv_47124381171730
// MI455X (gfx1250) — matrix operand order repaired and checked by reading, not compiled, not run on hardware
//
#include <hip/hip_runtime.h>

typedef __attribute__((ext_vector_type(16))) _Float16 v16h;
typedef __attribute__((ext_vector_type(8)))  _Float16 v8h;
typedef __attribute__((ext_vector_type(8)))  float    v8f;

#define N_PTS 32768      // N from reference setup
#define FSZ   7          // stencil width
#define HALFW 3
#define HDIM  128        // hidden width
#define RPW   16         // rows per wave (WMMA M)
#define WAVES 8
#define RPB   (RPW*WAVES)

__device__ __forceinline__ float elu_f(float x) {
    return x > 0.f ? x : (__expf(x) - 1.f);
}

// One 16x128 @ 128x128 layer step on the WMMA pipe, ELU fused in epilogue.
// hin : [16][128] f16 row-major (per-wave LDS)
// Wl  : [128][128] f16 row-major [out][in] (shared LDS)
// hout: [16][128] f16 row-major (per-wave LDS)
__device__ __forceinline__ void wmma_layer(const _Float16* __restrict__ hin,
                                           const _Float16* __restrict__ Wl,
                                           _Float16* __restrict__ hout,
                                           int lane) {
    const int sel = lane >> 4;   // 0: lanes 0-15, 1: lanes 16-31
    const int lr  = lane & 15;
    for (int nt = 0; nt < 8; ++nt) {           // 8 output col tiles of 16
        v8f acc = {};
        const int ocol = nt * 16 + lr;
        #pragma unroll
        for (int ks = 0; ks < 4; ++ks) {       // K = 128 in steps of 32
            const int kb = ks * 32;
            // A fragment: row lr, halves [kb+sel*8 .. +7] and [kb+16+sel*8 .. +7]
            v8h a_lo = *(const v8h*)(hin + lr * HDIM + kb + sel * 8);
            v8h a_hi = *(const v8h*)(hin + lr * HDIM + kb + 16 + sel * 8);
            v16h a = __builtin_shufflevector(a_lo, a_hi,
                                             0,1,2,3,4,5,6,7,8,9,10,11,12,13,14,15);
            // B fragment: col ocol, halves [kb+sel*8 .. +7] and [kb+16+sel*8 .. +7]
            v8h b_lo = *(const v8h*)(Wl + ocol * HDIM + kb + sel * 8);
            v8h b_hi = *(const v8h*)(Wl + ocol * HDIM + kb + 16 + sel * 8);
            v16h b = __builtin_shufflevector(b_lo, b_hi,
                                             0,1,2,3,4,5,6,7,8,9,10,11,12,13,14,15);
            acc = __builtin_amdgcn_wmma_f32_16x16x32_f16(
                      false, a, false, b, (short)0, acc, false, false);
        }
        // C layout: VGPR r -> row sel*8+r, col = nt*16 + lr. Fuse ELU + f16 cvt.
        #pragma unroll
        for (int r = 0; r < 8; ++r) {
            const int row = sel * 8 + r;
            hout[row * HDIM + ocol] = (_Float16)elu_f(acc[r]);
        }
    }
}

__global__ __launch_bounds__(256)
void mlpconv_stencil_kernel(const float* __restrict__ x,
                            const float* __restrict__ W0,
                            const float* __restrict__ W1,
                            const float* __restrict__ W2,
                            const float* __restrict__ W3,
                            float* __restrict__ out) {
    __shared__ __align__(32) _Float16 W1h[HDIM * HDIM];
    __shared__ __align__(32) _Float16 W2h[HDIM * HDIM];
    __shared__ float W0s[HDIM * FSZ];
    __shared__ float W3s[HDIM];
    __shared__ __align__(32) _Float16 hA[WAVES][RPW][HDIM];
    __shared__ __align__(32) _Float16 hB[WAVES][RPW][HDIM];
    __shared__ float xw[WAVES][RPW][8];

    const int tid = threadIdx.x;

    // ---- stage weights into LDS (f16 for WMMA layers, f32 for ends) ----
    for (int i = tid; i < HDIM * HDIM; i += 256) {
        W1h[i] = (_Float16)W1[i];
        W2h[i] = (_Float16)W2[i];
    }
    for (int i = tid; i < HDIM * FSZ; i += 256) W0s[i] = W0[i];
    if (tid < HDIM) W3s[tid] = W3[tid];
    __syncthreads();

    const int wave = tid >> 5;
    const int lane = tid & 31;
    const int rowbase = blockIdx.x * RPB + wave * RPW;

    // ---- gather periodic stencil windows for this wave's 16 rows ----
    if (lane < RPW) {
        const int r = rowbase + lane;
        const int b = r / N_PTS;
        const int n = r % N_PTS;
        const float* xb = x + (long)b * N_PTS;
        #pragma unroll
        for (int f = 0; f < FSZ; ++f) {
            int idx = n + f - HALFW;             // window order: roll +h .. -h
            idx = (idx < 0) ? idx + N_PTS : (idx >= N_PTS ? idx - N_PTS : idx);
            xw[wave][lane][f] = xb[idx];
        }
    }
    __syncthreads();

    // ---- layer 0: 7 -> 128 (VALU; ~450 MACs/lane) + ELU -> f16 in hA ----
    for (int m = 0; m < RPW; ++m) {
        float win[FSZ];
        #pragma unroll
        for (int f = 0; f < FSZ; ++f) win[f] = xw[wave][m][f];
        #pragma unroll
        for (int j = 0; j < 4; ++j) {
            const int o = lane + 32 * j;
            float acc = 0.f;
            #pragma unroll
            for (int f = 0; f < FSZ; ++f) acc += win[f] * W0s[o * FSZ + f];
            hA[wave][m][o] = (_Float16)elu_f(acc);
        }
    }
    __syncthreads();

    // ---- layer 1: 128 -> 128 on WMMA, ELU ----
    wmma_layer(&hA[wave][0][0], W1h, &hB[wave][0][0], lane);
    __syncthreads();

    // ---- layer 2: 128 -> 128 on WMMA, ELU ----
    wmma_layer(&hB[wave][0][0], W2h, &hA[wave][0][0], lane);
    __syncthreads();

    // ---- layer 3: 128 -> 1 dot product; lane pairs split K, shfl-combine ----
    {
        const int m   = lane & 15;
        const int sel = lane >> 4;
        const _Float16* h = &hA[wave][m][0];
        float acc = 0.f;
        #pragma unroll
        for (int k = 0; k < 64; ++k)
            acc += (float)h[sel * 64 + k] * W3s[sel * 64 + k];
        acc += __shfl_xor(acc, 16, 32);
        if (sel == 0) out[rowbase + m] = acc;
    }
}

extern "C" void kernel_launch(void* const* d_in, const int* in_sizes, int n_in,
                              void* d_out, int out_size, void* d_ws, size_t ws_size,
                              hipStream_t stream) {
    const float* x  = (const float*)d_in[0];
    const float* W0 = (const float*)d_in[1];
    const float* W1 = (const float*)d_in[2];
    const float* W2 = (const float*)d_in[3];
    const float* W3 = (const float*)d_in[4];
    float* out = (float*)d_out;

    const int M = in_sizes[0];        // B * N flattened rows
    const int blocks = M / RPB;       // 1,048,576 / 128 = 8192
    mlpconv_stencil_kernel<<<blocks, 256, 0, stream>>>(x, W0, W1, W2, W3, out);
}
